// TreeSimpleGRU_70600672412006
// MI455X (gfx1250) — matrix operand order repaired and checked by reading, not compiled, not run on hardware
//
#include <hip/hip_runtime.h>
#include <hip/hip_bf16.h>
#include <math.h>

#define MEM_DIM   150
#define IN_NODE   490
#define IN_LEAF   320
#define AT_HID    20
#define PS_STRIDE 480     // 450 GRU pre rows + 20 attention rows, padded
#define KSTAT     340     // static x columns: 300 word + 20 rel + 20 tag
#define KSTAT_PAD 352     // 11 k-slabs of 32
#define NB_ROWS   512     // padded B-operand row count (covers 470 and 450)

typedef __bf16 v16bf __attribute__((ext_vector_type(16)));
typedef __bf16 v8bf  __attribute__((ext_vector_type(8)));
typedef float  v8f   __attribute__((ext_vector_type(8)));

static __device__ __forceinline__ __bf16 f2bf(float f) {
  union { float f; unsigned u; } v; v.f = f;
  unsigned r = (v.u + 0x7FFFu + ((v.u >> 16) & 1u)) >> 16;   // round-nearest-even
  union { unsigned short s; __bf16 b; } o; o.s = (unsigned short)r;
  return o.b;
}

static __device__ __forceinline__ float sigmoidf_(float x) {
  return 1.0f / (1.0f + expf(-x));
}

// ---------------------------------------------------------------- utilities
__global__ void zero_kernel(float* p, long n) {
  long i = (long)blockIdx.x * blockDim.x + threadIdx.x;
  if (i < n) p[i] = 0.0f;
}

// uhb[r] = leaf_b[r] + sum_n leaf_U[r,n] * leaf_h[n]   (shared by all leaves)
__global__ void leaf_uhb_kernel(const float* __restrict__ U,
                                const float* __restrict__ b,
                                const float* __restrict__ h0,
                                float* __restrict__ uhb) {
  int r = blockIdx.x * blockDim.x + threadIdx.x;
  if (r < 3 * MEM_DIM) {
    float acc = b[r];
    const float* u = U + (long)r * MEM_DIM;
    for (int n = 0; n < MEM_DIM; ++n) acc += u[n] * h0[n];
    uhb[r] = acc;
  }
}

// ------------------------------------------------------------- pack kernels
// A operand for edge-static GEMM: [Epad x 352] bf16, row = edge
__global__ void pack_edge_A(const float* __restrict__ w_emb,
                            const float* __restrict__ rel_emb,
                            const float* __restrict__ tag_emb,
                            const int* __restrict__ edges,
                            __bf16* __restrict__ A, int E, long total) {
  long idx = (long)blockIdx.x * blockDim.x + threadIdx.x;
  if (idx >= total) return;
  int m = (int)(idx / KSTAT_PAD), k = (int)(idx % KSTAT_PAD);
  float v = 0.0f;
  if (m < E && k < KSTAT) {
    int p = edges[2 * m], c = edges[2 * m + 1];
    v = (k < 300) ? w_emb[(long)p * 300 + k]
      : (k < 320) ? rel_emb[(long)c * 20 + (k - 300)]
                  : tag_emb[(long)p * 20 + (k - 320)];
  }
  A[idx] = f2bf(v);
}

// B operand for edge-static GEMM: [512 x 352] bf16, row = output channel
// rows 0..449 = node_W (static cols), 450..469 = at_Wb, rest zero
__global__ void pack_edge_B(const float* __restrict__ node_W,
                            const float* __restrict__ at_Wb,
                            __bf16* __restrict__ B) {
  long idx = (long)blockIdx.x * blockDim.x + threadIdx.x;
  if (idx >= (long)NB_ROWS * KSTAT_PAD) return;
  int r = (int)(idx / KSTAT_PAD), k = (int)(idx % KSTAT_PAD);
  float v = 0.0f;
  if (k < KSTAT) {
    int col = (k < 300) ? k : (k + 150);        // skip dynamic K[c] cols 300..449
    if (r < 3 * MEM_DIM)                 v = node_W[(long)r * IN_NODE + col];
    else if (r < 3 * MEM_DIM + AT_HID)   v = at_Wb[(long)(r - 3 * MEM_DIM) * IN_NODE + col];
  }
  B[idx] = f2bf(v);
}

// A operand for leaf GEMM: [Lpad x 320] bf16
__global__ void pack_leaf_A(const float* __restrict__ w_emb,
                            const float* __restrict__ tag_emb,
                            const int* __restrict__ leaf_ids,
                            __bf16* __restrict__ A, int L, long total) {
  long idx = (long)blockIdx.x * blockDim.x + threadIdx.x;
  if (idx >= total) return;
  int m = (int)(idx / IN_LEAF), k = (int)(idx % IN_LEAF);
  float v = 0.0f;
  if (m < L) {
    int nid = leaf_ids[m];
    v = (k < 300) ? w_emb[(long)nid * 300 + k] : tag_emb[(long)nid * 20 + (k - 300)];
  }
  A[idx] = f2bf(v);
}

// B operand for leaf GEMM: [512 x 320] bf16, rows 0..449 = leaf_W
__global__ void pack_leaf_B(const float* __restrict__ leaf_W,
                            __bf16* __restrict__ B) {
  long idx = (long)blockIdx.x * blockDim.x + threadIdx.x;
  if (idx >= (long)NB_ROWS * IN_LEAF) return;
  int r = (int)(idx / IN_LEAF), k = (int)(idx % IN_LEAF);
  B[idx] = f2bf((r < 3 * MEM_DIM) ? leaf_W[(long)r * IN_LEAF + k] : 0.0f);
}

// ------------------------------------------ WMMA GEMM: C[M x N] = A * B^T
// A: [Mpad x K] bf16 row-major (padded, no guards needed)
// B: [512  x K] bf16 row-major (output channel rows, padded)
// one wave32 per 16x64 tile: A fragment reused across 4 WMMAs
__global__ void __launch_bounds__(32) gemm_bf16_kernel(
    const __bf16* __restrict__ A, const __bf16* __restrict__ B,
    float* __restrict__ C, int Mvalid, int Nvalid, int Kdim)
{
  const int lane = threadIdx.x;
  const int half = lane >> 4;                  // 0: lanes 0-15, 1: lanes 16-31
  const int l15  = lane & 15;
  const __bf16* arow = A + (size_t)(blockIdx.x * 16 + l15) * Kdim;
  const int ncol0 = blockIdx.y * 64;
  const __bf16* brow = B + (size_t)(ncol0 + l15) * Kdim;
  const size_t bstep = (size_t)16 * Kdim;
  v8f acc0 = {}, acc1 = {}, acc2 = {}, acc3 = {};
  for (int kk = 0; kk < Kdim; kk += 32) {
    // A fragment: one contiguous 16-element chunk, K = kk + half*16 + 0..15
    v16bf a = *(const v16bf*)(arow + kk + half * 16);
    // B fragment: one contiguous 16-element chunk, K = kk + half*16 + 0..15
    v16bf b0 = *(const v16bf*)(brow + kk + half * 16);
    v16bf b1 = *(const v16bf*)(brow + bstep + kk + half * 16);
    v16bf b2 = *(const v16bf*)(brow + 2 * bstep + kk + half * 16);
    v16bf b3 = *(const v16bf*)(brow + 3 * bstep + kk + half * 16);
    acc0 = __builtin_amdgcn_wmma_f32_16x16x32_bf16(false, a, false, b0, (short)0, acc0, false, false);
    acc1 = __builtin_amdgcn_wmma_f32_16x16x32_bf16(false, a, false, b1, (short)0, acc1, false, false);
    acc2 = __builtin_amdgcn_wmma_f32_16x16x32_bf16(false, a, false, b2, (short)0, acc2, false, false);
    acc3 = __builtin_amdgcn_wmma_f32_16x16x32_bf16(false, a, false, b3, (short)0, acc3, false, false);
  }
  // D layout: vgpr i -> M = i + half*8, N = l15
#pragma unroll
  for (int i = 0; i < 8; ++i) {
    int mm = blockIdx.x * 16 + i + half * 8;
    if (mm < Mvalid) {
      float* crow = C + (size_t)mm * PS_STRIDE;
      int n0 = ncol0 + l15;
      if (n0      < Nvalid) crow[n0]      = acc0[i];
      if (n0 + 16 < Nvalid) crow[n0 + 16] = acc1[i];
      if (n0 + 32 < Nvalid) crow[n0 + 32] = acc2[i];
      if (n0 + 48 < Nvalid) crow[n0 + 48] = acc3[i];
    }
  }
}

__global__ void leaf_fin_kernel(const float* __restrict__ Pl,
                                const float* __restrict__ uhb,
                                const float* __restrict__ h0,
                                const int* __restrict__ leaf_ids,
                                float* __restrict__ K, int L)
{
  int idx = blockIdx.x * blockDim.x + threadIdx.x;
  if (idx >= L * MEM_DIM) return;
  int l = idx / MEM_DIM, i = idx - l * MEM_DIM;
  const float* row = Pl + (long)l * PS_STRIDE;
  float z  = sigmoidf_(row[i]              + uhb[i]);
  float r  = sigmoidf_(row[MEM_DIM + i]    + uhb[MEM_DIM + i]);
  float ht = tanhf   (row[2 * MEM_DIM + i] + uhb[2 * MEM_DIM + i]);
  float h = h0[i];
  K[(long)leaf_ids[l] * MEM_DIM + i] = r * h + (1.0f - z) * ht;
}

// ----------------- residual serial scan: one persistent workgroup, LDS-resident h
__global__ void __launch_bounds__(512) scan_kernel(
    const float* __restrict__ node_W, const float* __restrict__ node_U,
    const float* __restrict__ node_b, const float* __restrict__ at_Wb,
    const float* __restrict__ at_bb,  const float* __restrict__ at_Wa,
    const float* __restrict__ at_ba,  const int* __restrict__ edges,
    const float* __restrict__ Ps, float* __restrict__ K,
    float* __restrict__ out, int E)
{
  __shared__ float xk[MEM_DIM];        // K[child]
  __shared__ float hb[MEM_DIM];        // running h for current parent
  __shared__ float pre[3 * MEM_DIM];
  __shared__ float att[AT_HID];
  const int tid = threadIdx.x;
  int prevp = -1;
  for (int e = 0; e < E; ++e) {
    const int p = edges[2 * e];
    const int c = edges[2 * e + 1];
    if (tid < MEM_DIM) {
      xk[tid] = K[(long)c * MEM_DIM + tid];
      if (p != prevp) hb[tid] = 0.0f;  // edges grouped by parent: fresh h
    }
    if (tid < 3 * MEM_DIM + AT_HID && e + 1 < E)
      __builtin_prefetch(&Ps[(long)(e + 1) * PS_STRIDE + tid], 0, 3);
    __syncthreads();
    if (tid < 3 * MEM_DIM) {
      float acc = Ps[(long)e * PS_STRIDE + tid] + node_b[tid];
      const float* w = node_W + (long)tid * IN_NODE + 300;  // dynamic K[c] cols
      const float* u = node_U + (long)tid * MEM_DIM;
#pragma unroll 5
      for (int i = 0; i < MEM_DIM; ++i) acc += w[i] * xk[i] + u[i] * hb[i];
      pre[tid] = acc;
    } else if (tid < 3 * MEM_DIM + AT_HID) {
      const int q = tid - 3 * MEM_DIM;
      float acc = Ps[(long)e * PS_STRIDE + tid] + at_bb[q];
      const float* w = at_Wb + (long)q * IN_NODE + 300;
#pragma unroll 5
      for (int i = 0; i < MEM_DIM; ++i) acc += w[i] * xk[i];
      att[q] = tanhf(acc);
    }
    __syncthreads();
    if (tid < MEM_DIM) {
      float s = at_ba[0];
#pragma unroll
      for (int j = 0; j < AT_HID; ++j) s += at_Wa[j] * att[j];
      const float a  = sigmoidf_(s);               // redundant per-thread: no barrier
      const float z  = sigmoidf_(pre[tid]);
      const float r  = sigmoidf_(pre[MEM_DIM + tid]);
      const float ht = tanhf   (pre[2 * MEM_DIM + tid]);
      const float h  = hb[tid];
      const float mm = r * h + (1.0f - z) * ht;
      const float hn = a * mm + (1.0f - a) * h;
      hb[tid] = hn;
      K[(long)p * MEM_DIM + tid] = hn;
    }
    prevp = p;
    __syncthreads();
  }
  if (tid < MEM_DIM) out[tid] = K[tid];            // root state K[0]
}

// ------------------------------------------------------------------ launcher
extern "C" void kernel_launch(void* const* d_in, const int* in_sizes, int n_in,
                              void* d_out, int out_size, void* d_ws, size_t ws_size,
                              hipStream_t stream)
{
  (void)n_in; (void)out_size; (void)ws_size;
  const float* w_emb   = (const float*)d_in[0];
  const float* tag_emb = (const float*)d_in[1];
  const float* rel_emb = (const float*)d_in[2];
  const float* leaf_h  = (const float*)d_in[3];
  const float* leaf_W  = (const float*)d_in[4];
  const float* leaf_U  = (const float*)d_in[5];
  const float* leaf_b  = (const float*)d_in[6];
  const float* node_W  = (const float*)d_in[7];
  const float* node_U  = (const float*)d_in[8];
  const float* node_b  = (const float*)d_in[9];
  const float* at_Wb   = (const float*)d_in[10];
  const float* at_bb   = (const float*)d_in[11];
  const float* at_Wa   = (const float*)d_in[12];
  const float* at_ba   = (const float*)d_in[13];
  const int*   edges   = (const int*)d_in[14];
  const int*   leaf_ids= (const int*)d_in[15];

  const int N = in_sizes[0] / 300;
  const int E = in_sizes[14] / 2;
  const int L = in_sizes[15];
  const int Epad = (E + 15) & ~15;
  const int Lpad = (L + 15) & ~15;

  // ---- workspace carve-up (256B-aligned regions; leaf & edge phases reuse) ----
  char* base = (char*)d_ws;
  size_t off = 0;
  auto carve = [&](size_t bytes) -> void* {
    void* p = base + off;
    off += (bytes + 255) & ~(size_t)255;
    return p;
  };
  float*  Kbuf = (float*)carve((size_t)N * MEM_DIM * sizeof(float));
  float*  P    = (float*)carve((size_t)Epad * PS_STRIDE * sizeof(float));  // Pl then Ps
  float*  uhb  = (float*)carve(PS_STRIDE * sizeof(float));
  size_t  aElems = (size_t)Epad * KSTAT_PAD;
  size_t  aLeaf  = (size_t)Lpad * IN_LEAF;
  __bf16* Abuf = (__bf16*)carve((aElems > aLeaf ? aElems : aLeaf) * sizeof(__bf16));
  __bf16* Bbuf = (__bf16*)carve((size_t)NB_ROWS * KSTAT_PAD * sizeof(__bf16));

  long nK = (long)N * MEM_DIM;
  zero_kernel<<<(int)((nK + 255) / 256), 256, 0, stream>>>(Kbuf, nK);
  leaf_uhb_kernel<<<1, 512, 0, stream>>>(leaf_U, leaf_b, leaf_h, uhb);

  // ---- leaf phase: pack -> WMMA GEMM -> activations/scatter ----
  long tLA = (long)Lpad * IN_LEAF;
  pack_leaf_A<<<(int)((tLA + 255) / 256), 256, 0, stream>>>(
      w_emb, tag_emb, leaf_ids, Abuf, L, tLA);
  long tLB = (long)NB_ROWS * IN_LEAF;
  pack_leaf_B<<<(int)((tLB + 255) / 256), 256, 0, stream>>>(leaf_W, Bbuf);
  dim3 lgrid(Lpad / 16, 8);                               // 16x64 tiles, 512 cols
  gemm_bf16_kernel<<<lgrid, 32, 0, stream>>>(Abuf, Bbuf, P, L, 3 * MEM_DIM, IN_LEAF);
  leaf_fin_kernel<<<(L * MEM_DIM + 255) / 256, 256, 0, stream>>>(
      P, uhb, leaf_h, leaf_ids, Kbuf, L);

  // ---- edge-static phase: pack -> WMMA GEMM (reuses Abuf/Bbuf/P) ----
  long tEA = (long)Epad * KSTAT_PAD;
  pack_edge_A<<<(int)((tEA + 255) / 256), 256, 0, stream>>>(
      w_emb, rel_emb, tag_emb, edges, Abuf, E, tEA);
  long tEB = (long)NB_ROWS * KSTAT_PAD;
  pack_edge_B<<<(int)((tEB + 255) / 256), 256, 0, stream>>>(node_W, at_Wb, Bbuf);
  dim3 egrid(Epad / 16, 8);
  gemm_bf16_kernel<<<egrid, 32, 0, stream>>>(Abuf, Bbuf, P, E, PS_STRIDE, KSTAT_PAD);

  // ---- residual serial scan ----
  scan_kernel<<<1, 512, 0, stream>>>(node_W, node_U, node_b, at_Wb, at_bb,
                                     at_Wa, at_ba, edges, P, Kbuf,
                                     (float*)d_out, E);
}
